// SeparateFastDctRNN_85899345920340
// MI455X (gfx1250) — matrix operand order repaired and checked by reading, not compiled, not run on hardware
//
#include <hip/hip_runtime.h>
#include <hip/hip_bf16.h>

// Problem constants
#define TT   32
#define BB   16
#define II   512
#define HH   512
#define NCC  2556     // number of triu coefficients
#define NCP  2560     // NC padded to multiple of 32
#define NTH  160      // NCP/16  (N-tiles per gate)
#define KTX  16       // II/32   (K-tiles for x-part)
#define KTH  80       // NCP/32  (K-tiles for h-part)
#define NROW 71       // nonzero rows of sparse S
#define NTP  5        // 80/16 N-tiles for P GEMM (need cols 0..70)

typedef __attribute__((ext_vector_type(16))) __bf16 v16bf;
typedef __attribute__((ext_vector_type(8)))  float  v8f;

__device__ __forceinline__ unsigned short f2bf(float f) {
    unsigned u = __float_as_uint(f);
    unsigned r = u + 0x7FFFu + ((u >> 16) & 1u);   // round-to-nearest-even
    return (unsigned short)(r >> 16);
}
__device__ __forceinline__ float sigm(float x) { return 1.0f / (1.0f + __expf(-x)); }
__device__ __forceinline__ v16bf ldt(const unsigned short* p) {
    return *reinterpret_cast<const v16bf*>(p);
}
__device__ __forceinline__ v8f wmma_bf16(v16bf a, v16bf b, v8f c) {
    return __builtin_amdgcn_wmma_f32_16x16x32_bf16(false, a, false, b, (short)0, c, false, false);
}
// A-layout (16x32 bf16, ISA 7.12.2): lane = lh*16+m ; element e -> klocal
__device__ __forceinline__ int a_klocal(int lh, int e) { return e + lh * 8 + (e >= 8 ? 8 : 0); }
// inverse: klocal -> (lh, e)
__device__ __forceinline__ void a_inv(int kl, int& lh, int& e) {
    lh = (kl >> 3) & 1;
    e  = (kl & 7) + ((kl & 16) ? 8 : 0);
}

// ---------------- packing kernels ----------------

// input_ (T*B=512 rows, I=512) fp32 -> A tiles [Mt=32][Kt=16][32 lanes][16 el]
__global__ void pack_a_kernel(const float* __restrict__ X, unsigned short* __restrict__ out) {
    int idx = blockIdx.x * blockDim.x + threadIdx.x;          // 32*16*512 threads
    int e  = idx & 15;
    int L  = (idx >> 4) & 31;
    int tl = idx >> 9;
    int kt = tl % KTX;
    int mt = tl / KTX;
    int m  = mt * 16 + (L & 15);
    int k  = kt * 32 + a_klocal(L >> 4, e);
    out[idx] = f2bf(X[(size_t)m * II + k]);
}

// LSTM weight (4*NC rows, Kv cols) fp32 -> B tiles [640][Kt][32][16]
// B[k][n] = W[n][k]; gate-padded columns: n -> g = n/NCP, nc = n%NCP, row = g*NC+nc
__global__ void pack_w_kernel(const float* __restrict__ W, unsigned short* __restrict__ out,
                              int Kt, int Kv) {
    long long idx = (long long)blockIdx.x * blockDim.x + threadIdx.x; // 640*Kt*512
    int e  = (int)(idx & 15);
    int L  = (int)((idx >> 4) & 31);
    long long tl = idx >> 9;
    int kt = (int)(tl % Kt);
    int ng = (int)(tl / Kt);
    int n  = ng * 16 + (L & 15);
    int g  = n / NCP, nc = n % NCP;
    int k  = kt * 32 + a_klocal(L >> 4, e);                   // B-layout: same k order as A
    float v = 0.0f;
    if (nc < NCC && k < Kv) v = W[(size_t)(g * NCC + nc) * Kv + k];
    out[idx] = f2bf(v);
}

// idct matrix (512x512) fp32 -> B tiles [5][16][32][16], B[k][n] = M[k][n], n < 80
__global__ void pack_idct_kernel(const float* __restrict__ M, unsigned short* __restrict__ out) {
    int idx = blockIdx.x * blockDim.x + threadIdx.x;          // 5*16*512 threads
    int e  = idx & 15;
    int L  = (idx >> 4) & 31;
    int tl = idx >> 9;
    int kt = tl % KTX;
    int nt = tl / KTX;
    int n  = nt * 16 + (L & 15);
    int k  = kt * 32 + a_klocal(L >> 4, e);
    out[idx] = f2bf(M[(size_t)k * II + n]);
}

__global__ void zero_kernel(unsigned int* __restrict__ p, int nwords) {
    int i = blockIdx.x * blockDim.x + threadIdx.x;
    if (i < nwords) p[i] = 0u;
}

// ---------------- phase 2: fused LSTM step (both LSTMs) ----------------
// grid (160, 2), block 32 (one wave). Each wave owns 16 hidden units across all 4 gates.
__global__ void lstm_step_kernel(int t,
        const unsigned short* __restrict__ Ain,
        const unsigned short* __restrict__ Wx1, const unsigned short* __restrict__ Wh1,
        const float* __restrict__ b1, const float* __restrict__ sc1,
        float* __restrict__ c1, const unsigned short* __restrict__ hr1,
        unsigned short* __restrict__ hw1, float* __restrict__ co1,
        const unsigned short* __restrict__ Wx2, const unsigned short* __restrict__ Wh2,
        const float* __restrict__ b2, const float* __restrict__ sc2,
        float* __restrict__ c2, const unsigned short* __restrict__ hr2,
        unsigned short* __restrict__ hw2, float* __restrict__ co2) {
    const int nt  = blockIdx.x;
    const int lid = threadIdx.x;
    const bool l2 = (blockIdx.y != 0);
    const unsigned short* Wx = l2 ? Wx2 : Wx1;
    const unsigned short* Wh = l2 ? Wh2 : Wh1;
    const float* bias = l2 ? b2 : b1;
    const float* scl  = l2 ? sc2 : sc1;
    float* cst        = l2 ? c2 : c1;
    const unsigned short* hr = l2 ? hr2 : hr1;
    unsigned short* hw       = l2 ? hw2 : hw1;
    float* co                = l2 ? co2 : co1;

    const int n16 = lid & 15, lh = lid >> 4;
    const int nc  = nt * 16 + n16;

    v8f acc[4];
    #pragma unroll
    for (int g = 0; g < 4; ++g) {
        float bv = (nc < NCC) ? bias[g * NCC + nc] : 0.0f;
        #pragma unroll
        for (int r = 0; r < 8; ++r) acc[g][r] = bv;
    }

    // x @ Wx^T  (K = 512)
    for (int kt = 0; kt < KTX; ++kt) {
        v16bf a = ldt(Ain + (((size_t)(t * 16 + kt)) * 32 + lid) * 16);
        #pragma unroll
        for (int g = 0; g < 4; ++g) {
            v16bf b = ldt(Wx + ((((size_t)(g * NTH + nt)) * KTX + kt) * 32 + lid) * 16);
            acc[g] = wmma_bf16(a, b, acc[g]);
        }
    }
    // h @ Wh^T  (K = 2560)
    for (int kt = 0; kt < KTH; ++kt) {
        if (kt + 1 < KTH)
            __builtin_prefetch(Wh + ((((size_t)nt) * KTH + kt + 1) * 32 + lid) * 16, 0, 3);
        v16bf a = ldt(hr + ((size_t)kt * 32 + lid) * 16);
        #pragma unroll
        for (int g = 0; g < 4; ++g) {
            v16bf b = ldt(Wh + ((((size_t)(g * NTH + nt)) * KTH + kt) * 32 + lid) * 16);
            acc[g] = wmma_bf16(a, b, acc[g]);
        }
    }

    // fused gate epilogue; C/D layout: lane n = lid&15, rows m = lh*8 + r
    const float sv = (nc < NCC) ? scl[nc] : 0.0f;
    int kt2 = nc >> 5, kl = nc & 31, lh2, e2;
    a_inv(kl, lh2, e2);
    #pragma unroll
    for (int r = 0; r < 8; ++r) {
        int m = lh * 8 + r;
        float gi = acc[0][r], gf = acc[1][r], gg = acc[2][r], go = acc[3][r];
        float cold = cst[(size_t)m * NCP + nc];
        float cnew = sigm(gf) * cold + sigm(gi) * tanhf(gg);
        float h    = sigm(go) * tanhf(cnew);
        cst[(size_t)m * NCP + nc] = cnew;
        if (nc < NCC)
            co[((size_t)(t * BB + m)) * NCC + nc] = h * sv;
        hw[(((size_t)kt2 * 32) + (lh2 * 16 + m)) * 16 + e2] = f2bf(h);   // repack A-layout
    }
}

// ---------------- phase 3a: P = X @ idct (only cols 0..79 needed) ----------------
// grid (5, 2), block 32. id 0: X = x_t ; id 1: X = h_prev
__global__ void p_kernel(int t,
        const unsigned short* __restrict__ Ain, const unsigned short* __restrict__ h3p,
        const unsigned short* __restrict__ idctIp, const unsigned short* __restrict__ idctHp,
        float* __restrict__ P) {
    const int nt  = blockIdx.x;
    const int id  = blockIdx.y;
    const int lid = threadIdx.x;
    const unsigned short* Bp = id ? idctHp : idctIp;
    v8f acc;
    #pragma unroll
    for (int r = 0; r < 8; ++r) acc[r] = 0.0f;
    for (int kt = 0; kt < KTX; ++kt) {
        v16bf a = id ? ldt(h3p + ((size_t)kt * 32 + lid) * 16)
                     : ldt(Ain + (((size_t)(t * 16 + kt)) * 32 + lid) * 16);
        v16bf b = ldt(Bp + ((((size_t)nt) * KTX + kt) * 32 + lid) * 16);
        acc = wmma_bf16(a, b, acc);
    }
    const int n = nt * 16 + (lid & 15), lh = lid >> 4;
    #pragma unroll
    for (int r = 0; r < 8; ++r) {
        int m = lh * 8 + r;
        P[(size_t)id * BB * 80 + (size_t)m * 80 + n] = acc[r];
    }
}

// ---------------- phase 3b: sparse z, u = idct_hid[:, :71] @ z, tanh ----------------
// grid (16 = batch), block 512
__global__ void k2_kernel(int t,
        const float* __restrict__ co1, const float* __restrict__ co2,
        const float* __restrict__ P,   const float* __restrict__ bias,
        const float* __restrict__ idct_hid,
        float* __restrict__ out, unsigned short* __restrict__ h3p) {
    __shared__ float z[NROW];
    const int b = blockIdx.x, tid = threadIdx.x;
    if (tid < NROW) {
        const int r   = tid;
        const int off = r * NROW - (r * (r - 1)) / 2;   // row start in coeff list
        const int len = NROW - r;
        const float* ci = co1 + ((size_t)(t * BB + b)) * NCC + off;
        const float* ch = co2 + ((size_t)(t * BB + b)) * NCC + off;
        const float* Pi = P + (size_t)b * 80;
        const float* Ph = P + (size_t)BB * 80 + (size_t)b * 80;
        float acc = 0.0f;
        for (int j = 0; j < len; ++j) {                 // col = r+441+j ; I-1-col = 70-r-j
            int pidx = (NROW - 1) - r - j;
            acc += ci[j] * Pi[pidx] + ch[j] * Ph[pidx];
        }
        z[r] = acc;
    }
    __syncthreads();
    const int hidx = tid;
    float u = bias[hidx];
    const float* Bm = idct_hid + (size_t)hidx * HH;
    #pragma unroll 8
    for (int r = 0; r < NROW; ++r) u += Bm[r] * z[r];
    float hn = tanhf(u);
    out[((size_t)(t * BB + b)) * HH + hidx] = hn;
    int kt = hidx >> 5, kl = hidx & 31, lh2, e2;
    a_inv(kl, lh2, e2);
    h3p[((size_t)kt * 32 + (lh2 * 16 + b)) * 16 + e2] = f2bf(hn);
}

// ---------------- launcher ----------------
extern "C" void kernel_launch(void* const* d_in, const int* in_sizes, int n_in,
                              void* d_out, int out_size, void* d_ws, size_t ws_size,
                              hipStream_t stream) {
    (void)in_sizes; (void)n_in; (void)out_size; (void)ws_size;
    const float* input_   = (const float*)d_in[0];
    const float* wih_ih   = (const float*)d_in[1];
    const float* wih_hh   = (const float*)d_in[2];
    const float* wih_b    = (const float*)d_in[3];
    const float* whh_ih   = (const float*)d_in[4];
    const float* whh_hh   = (const float*)d_in[5];
    const float* whh_b    = (const float*)d_in[6];
    const float* sc_ih    = (const float*)d_in[7];
    const float* sc_hh    = (const float*)d_in[8];
    const float* bias     = (const float*)d_in[9];
    const float* idct_in  = (const float*)d_in[10];
    const float* idct_hid = (const float*)d_in[11];
    float* out = (float*)d_out;
    char* ws = (char*)d_ws;

    size_t off = 0;
    auto alloc = [&](size_t bytes) { size_t o = off; off = (off + bytes + 511) & ~(size_t)511; return o; };
    const size_t AIN_E  = (size_t)32 * KTX * 32 * 16;          // input A tiles
    const size_t IDCT_E = (size_t)NTP * KTX * 32 * 16;
    const size_t WX_E   = (size_t)640 * KTX * 32 * 16;
    const size_t WH_E   = (size_t)640 * KTH * 32 * 16;
    const size_t CO_E   = (size_t)TT * BB * NCC;
    const size_t HP_E   = (size_t)KTH * 32 * 16;               // one h buffer (2560x16 bf16)
    const size_t H3_E   = (size_t)KTX * 32 * 16;               // 512x16 bf16

    unsigned short* Ain    = (unsigned short*)(ws + alloc(AIN_E * 2));
    unsigned short* idctIp = (unsigned short*)(ws + alloc(IDCT_E * 2));
    unsigned short* idctHp = (unsigned short*)(ws + alloc(IDCT_E * 2));
    unsigned short* Wx1p   = (unsigned short*)(ws + alloc(WX_E * 2));
    unsigned short* Wx2p   = (unsigned short*)(ws + alloc(WX_E * 2));
    unsigned short* Wh1p   = (unsigned short*)(ws + alloc(WH_E * 2));
    unsigned short* Wh2p   = (unsigned short*)(ws + alloc(WH_E * 2));
    float*          co1    = (float*)(ws + alloc(CO_E * 4));
    float*          co2    = (float*)(ws + alloc(CO_E * 4));
    float*          Pbuf   = (float*)(ws + alloc((size_t)2 * BB * 80 * 4));
    // zero-initialized state region (contiguous)
    size_t zoff = off;
    float*          c1     = (float*)(ws + alloc((size_t)BB * NCP * 4));
    float*          c2     = (float*)(ws + alloc((size_t)BB * NCP * 4));
    unsigned short* hb1    = (unsigned short*)(ws + alloc(2 * HP_E * 2));
    unsigned short* hb2    = (unsigned short*)(ws + alloc(2 * HP_E * 2));
    unsigned short* h3p    = (unsigned short*)(ws + alloc(H3_E * 2));
    size_t zwords = (off - zoff) / 4;

    zero_kernel<<<(int)((zwords + 255) / 256), 256, 0, stream>>>((unsigned int*)(ws + zoff), (int)zwords);
    pack_a_kernel<<<(int)(AIN_E / 256), 256, 0, stream>>>(input_, Ain);
    pack_idct_kernel<<<(int)(IDCT_E / 256), 256, 0, stream>>>(idct_in, idctIp);
    pack_idct_kernel<<<(int)(IDCT_E / 256), 256, 0, stream>>>(idct_hid, idctHp);
    pack_w_kernel<<<(int)(WX_E / 256), 256, 0, stream>>>(wih_ih, Wx1p, KTX, II);
    pack_w_kernel<<<(int)(WX_E / 256), 256, 0, stream>>>(whh_ih, Wx2p, KTX, II);
    pack_w_kernel<<<(int)(WH_E / 256), 256, 0, stream>>>(wih_hh, Wh1p, KTH, NCC);
    pack_w_kernel<<<(int)(WH_E / 256), 256, 0, stream>>>(whh_hh, Wh2p, KTH, NCC);

    // phase 2: both LSTMs, double-buffered h across t
    for (int t = 0; t < TT; ++t) {
        const unsigned short* hr1 = hb1 + (size_t)(t & 1) * HP_E;
        unsigned short*       hw1 = hb1 + (size_t)((t + 1) & 1) * HP_E;
        const unsigned short* hr2 = hb2 + (size_t)(t & 1) * HP_E;
        unsigned short*       hw2 = hb2 + (size_t)((t + 1) & 1) * HP_E;
        lstm_step_kernel<<<dim3(NTH, 2), 32, 0, stream>>>(t, Ain,
            Wx1p, Wh1p, wih_b, sc_ih, c1, hr1, hw1, co1,
            Wx2p, Wh2p, whh_b, sc_hh, c2, hr2, hw2, co2);
    }
    // phase 3: factored hyper-network recurrence
    for (int t = 0; t < TT; ++t) {
        p_kernel<<<dim3(NTP, 2), 32, 0, stream>>>(t, Ain, h3p, idctIp, idctHp, Pbuf);
        k2_kernel<<<BB, HH, 0, stream>>>(t, co1, co2, Pbuf, bias, idct_hid, out, h3p);
    }
}
